// PolyVGG16_85203561218728
// MI455X (gfx1250) — matrix operand order repaired and checked by reading, not compiled, not run on hardware
//
#include <hip/hip_runtime.h>
#include <cstdint>
#include <cstddef>

// ---------------------------------------------------------------------------
// PolyKervNet VGG16 forward for MI455X (gfx1250): implicit-GEMM convs via
// v_wmma_f32_16x16x32_bf16 with async global->LDS staging (ASYNCcnt pipelined,
// barrier-free: each wave stages and consumes its own LDS rows).
// ---------------------------------------------------------------------------

typedef __bf16 bf16;
typedef __attribute__((ext_vector_type(4)))  bf16  v4bf;
typedef __attribute__((ext_vector_type(8)))  bf16  v8bf;
typedef __attribute__((ext_vector_type(16))) bf16  v16bf;
typedef __attribute__((ext_vector_type(8)))  float v8f;

#define BATCH 256
#define NUM_CLASSES 100

// --------------------------------------------------------- async LDS helpers
__device__ __forceinline__ void async_b128_to_lds(unsigned ldsOff, const void* gsrc) {
  asm volatile("global_load_async_to_lds_b128 %0, %1, off"
               :: "v"(ldsOff), "v"(gsrc) : "memory");
}
__device__ __forceinline__ void wait_async_zero() {
  asm volatile("s_wait_asynccnt 0" ::: "memory");
}

// ---------------------------------------------------------------- utilities
__global__ void zero_words_kernel(uint32_t* __restrict__ p, long nwords) {
  long i = (long)blockIdx.x * blockDim.x + threadIdx.x;
  long stride = (long)gridDim.x * blockDim.x;
  for (; i < nwords; i += stride) p[i] = 0u;
}

// x[256][3][32][32] f32 -> padded channels-last bf16 [256][34][34][32]
__global__ void pack_input_kernel(const float* __restrict__ x, bf16* __restrict__ out) {
  const long total = (long)BATCH * 34 * 34 * 32;
  long i = (long)blockIdx.x * blockDim.x + threadIdx.x;
  const long stride = (long)gridDim.x * blockDim.x;
  for (; i < total; i += stride) {
    int c  = (int)(i & 31);
    long t = i >> 5;
    int xp = (int)(t % 34); t /= 34;
    int yp = (int)(t % 34); t /= 34;
    int b  = (int)t;
    float v = 0.0f;
    if (c < 3 && yp >= 1 && yp <= 32 && xp >= 1 && xp <= 32)
      v = x[(((long)b * 3 + c) * 32 + (yp - 1)) * 32 + (xp - 1)];
    out[i] = (bf16)v;
  }
}

// w[Cout][Cin][3][3] f32 -> bf16 [9*CinPad/32][Cout][32] (B-fragment friendly)
__global__ void pack_weight_kernel(const float* __restrict__ w, bf16* __restrict__ out,
                                   int Cin, int CinPad, int Cout) {
  const long total = (long)Cout * CinPad * 9;
  long i = (long)blockIdx.x * blockDim.x + threadIdx.x;
  const long stride = (long)gridDim.x * blockDim.x;
  const int chunksC = CinPad >> 5;
  for (; i < total; i += stride) {
    int kin = (int)(i & 31);
    long t = i >> 5;
    int n = (int)(t % Cout); t /= Cout;
    int chunk = (int)t;                 // 0 .. 9*chunksC-1
    int ct = chunk % chunksC;
    int kpos = chunk / chunksC;
    int ky = kpos / 3, kx = kpos - ky * 3;
    int ic = ct * 32 + kin;
    float v = 0.0f;
    if (ic < Cin)
      v = w[(((long)n * Cin + ic) * 3 + ky) * 3 + kx];
    out[i] = (bf16)v;
  }
}

// --------------------------------------------------------------- conv + poly
// act : padded input  [B][H+2][W+2][CinPad] bf16 (zero halo)
// wt  : packed weights [9*CinPad/32][Cout][32] bf16
// out : padded output [B][H+2][W+2][Cout]  bf16 (halo pre-zeroed)
//
// Block tile 128M x 64N, 8 waves. Wave w owns M-rows [16w, 16w+16) and all
// 64 N columns (4 accumulators). Staging map: thread t stages row t>>1, so
// each wave stages exactly the rows it reads -> no workgroup barriers; the
// async copy is synchronized per-wave with s_wait_asynccnt, double-buffered.
__global__ void kerv2d_wmma_kernel(const bf16* __restrict__ act,
                                   const bf16* __restrict__ wt,
                                   const float* __restrict__ bias,
                                   bf16* __restrict__ out,
                                   int H, int W, int CinPad, int Cout) {
  __shared__ bf16 As[2][128][40];             // double-buffered 128M x 32K, +8 pad

  const int tid  = threadIdx.x;
  const int lane = tid & 31;
  const int wid  = tid >> 5;                  // 0..7 -> M subtile

  const int mBlock = blockIdx.x * 128;
  const int nBlock = blockIdx.y * 64;
  const int Hp = H + 2, Wp = W + 2;
  const int HW = H * W;
  const int chunksC = CinPad >> 5;
  const int nSteps = 9 * chunksC;

  // --- staging coordinates: thread stages 32B (16 halfs) of one LDS row ---
  const int sRow  = tid >> 1;                 // 0..127
  const int sHalf = tid & 1;                  // half-select: halfs [0,16) / [16,32)
  const int mS = mBlock + sRow;
  const int bS = mS / HW;
  const int rS = mS - bS * HW;
  const int yS = rS / W;
  const int xS = rS - yS * W;
  // base pointer at tap (ky=0,kx=0), chunk 0, incl. half-select
  const bf16* p00 = act + ((long)(bS * Hp + yS) * Wp + xS) * CinPad + sHalf * 16;
  const unsigned ldsStageBase =
      (unsigned)(size_t)(&As[0][0][0]) + (unsigned)((sRow * 40 + sHalf * 16) * 2);

  // --- fragment lane mapping (wave32): contiguous k = 16*(lane/16) + s, A and B alike ---
  const int fragRow = wid * 16 + (lane & 15);         // A: M row within block
  const int kHalfA  = (lane < 16) ? 0 : 16;           // A: K base 0/16, 16 contig halfs
  const int kBaseB  = (lane < 16) ? 0 : 16;           // B: K base 0/16, 16 contig halfs
  const int nCol0   = nBlock + (lane & 15);

  v8f acc0 = {}, acc1 = {}, acc2 = {}, acc3 = {};

  // --- prologue: stage chunk 0 (ky=0,kx=0,ct=0) into LDS buffer 0 ---
  async_b128_to_lds(ldsStageBase,      (const void*)p00);
  async_b128_to_lds(ldsStageBase + 16, (const void*)(p00 + 8));

  int ky = 0, kx = 0, ct = 0;                 // indices of chunk just staged
  for (int step = 0; step < nSteps; ++step) {
    wait_async_zero();                        // current buffer's DMA complete

    // --- issue async stage of next chunk into the other LDS buffer ---
    if (step + 1 < nSteps) {
      int nct = ct + 1, nky = ky, nkx = kx;
      if (nct == chunksC) { nct = 0; ++nkx; if (nkx == 3) { nkx = 0; ++nky; } }
      const bf16* src = p00 + ((long)(nky * Wp + nkx)) * CinPad + nct * 32;
      const unsigned l0 = ldsStageBase + (unsigned)(((step + 1) & 1) * 128 * 40 * 2);
      async_b128_to_lds(l0,      (const void*)src);
      async_b128_to_lds(l0 + 16, (const void*)(src + 8));
      ky = nky; kx = nkx; ct = nct;
    }

    // --- A fragment: two 16B LDS reads per lane (wave-private rows) ---
    const bf16* asBuf = &As[step & 1][0][0];
    v8bf alo = *(const v8bf*)(asBuf + fragRow * 40 + kHalfA);
    v8bf ahi = *(const v8bf*)(asBuf + fragRow * 40 + kHalfA + 8);
    v16bf afrag = __builtin_shufflevector(alo, ahi,
        0, 1, 2, 3, 4, 5, 6, 7, 8, 9, 10, 11, 12, 13, 14, 15);

    // --- 4 B fragments: contiguous 32B per lane, n-subtiles 0..3 ---
    const bf16* wp = wt + ((long)step * Cout + nCol0) * 32 + kBaseB;
    v16bf b0 = *(const v16bf*)(wp);
    v16bf b1 = *(const v16bf*)(wp + 16 * 32);
    v16bf b2 = *(const v16bf*)(wp + 32 * 32);
    v16bf b3 = *(const v16bf*)(wp + 48 * 32);
    __builtin_prefetch((const char*)wp + (size_t)Cout * 64, 0, 1);  // next K chunk

    acc0 = __builtin_amdgcn_wmma_f32_16x16x32_bf16(false, afrag, false, b0,
                                                   (short)0, acc0, false, false);
    acc1 = __builtin_amdgcn_wmma_f32_16x16x32_bf16(false, afrag, false, b1,
                                                   (short)0, acc1, false, false);
    acc2 = __builtin_amdgcn_wmma_f32_16x16x32_bf16(false, afrag, false, b2,
                                                   (short)0, acc2, false, false);
    acc3 = __builtin_amdgcn_wmma_f32_16x16x32_bf16(false, afrag, false, b3,
                                                   (short)0, acc3, false, false);
  }

  // --- epilogue: (y+1)^2 + bias, bf16 store to interior of padded output ---
  const float bv0 = bias[nCol0];
  const float bv1 = bias[nCol0 + 16];
  const float bv2 = bias[nCol0 + 32];
  const float bv3 = bias[nCol0 + 48];
  const int mLaneOfs = (lane < 16) ? 0 : 8;
#pragma unroll
  for (int r = 0; r < 8; ++r) {
    const int m = mBlock + wid * 16 + r + mLaneOfs;
    const int bI = m / HW;
    const int rm = m - bI * HW;
    const int yy = rm / W;
    const int xx = rm - yy * W;
    bf16* po = out + (((long)bI * Hp + (yy + 1)) * Wp + (xx + 1)) * Cout + nCol0;
    float y0 = acc0[r] + 1.0f, y1 = acc1[r] + 1.0f;
    float y2 = acc2[r] + 1.0f, y3 = acc3[r] + 1.0f;
    po[0]  = (bf16)(y0 * y0 + bv0);
    po[16] = (bf16)(y1 * y1 + bv1);
    po[32] = (bf16)(y2 * y2 + bv2);
    po[48] = (bf16)(y3 * y3 + bv3);
  }
}

// ------------------------------------------------------------------ avgpool
// in : padded [B][Hin+2][Hin+2][C] -> out: padded [B][Hin/2+2][Hin/2+2][C]
__global__ void avgpool2_kernel(const bf16* __restrict__ in, bf16* __restrict__ out,
                                int C, int Hin) {
  const int Ho = Hin >> 1;
  const int Wip = Hin + 2, Wop = Ho + 2;
  const long total = (long)BATCH * Wop * Wop * C;
  long i = (long)blockIdx.x * blockDim.x + threadIdx.x;
  const long stride = (long)gridDim.x * blockDim.x;
  for (; i < total; i += stride) {
    int c  = (int)(i % C);
    long t = i / C;
    int xp = (int)(t % Wop); t /= Wop;
    int yp = (int)(t % Wop);
    int b  = (int)(t / Wop);
    float v = 0.0f;
    if (yp >= 1 && yp <= Ho && xp >= 1 && xp <= Ho) {
      const int y0 = (yp - 1) * 2 + 1, x0 = (xp - 1) * 2 + 1;
      const bf16* p = in + (((long)b * Wip + y0) * Wip + x0) * C + c;
      const long rs = (long)Wip * C;
      v = 0.25f * ((float)p[0] + (float)p[C] + (float)p[rs] + (float)p[rs + C]);
    }
    out[i] = (bf16)v;
  }
}

// ---------------------------------------------------------------------- FC
// act: padded [256][3][3][512] bf16 (interior at (1,1)); out: f32 [256][100]
__global__ void fc_kernel(const bf16* __restrict__ act, const float* __restrict__ wfc,
                          const float* __restrict__ bfc, float* __restrict__ out) {
  const int idx = blockIdx.x * blockDim.x + threadIdx.x;
  if (idx >= BATCH * NUM_CLASSES) return;
  const int j = idx % NUM_CLASSES;
  const int b = idx / NUM_CLASSES;
  const bf16* a = act + (((long)b * 3 + 1) * 3 + 1) * 512;
  float s = bfc[j];
  const float* wr = wfc + (long)j * 512;
#pragma unroll 8
  for (int c = 0; c < 512; ++c) s += (float)a[c] * wr[c];
  out[idx] = s;
}

// --------------------------------------------------------------------- host
extern "C" void kernel_launch(void* const* d_in, const int* in_sizes, int n_in,
                              void* d_out, int out_size, void* d_ws, size_t ws_size,
                              hipStream_t stream) {
  (void)in_sizes; (void)n_in; (void)out_size; (void)ws_size;

  struct LCfg { int cin, cinPad, cout, h; };
  static const LCfg L[13] = {
      {  3,  32,  64, 32}, { 64,  64,  64, 32},
      { 64,  64, 128, 16}, {128, 128, 128, 16},
      {128, 128, 256,  8}, {256, 256, 256,  8}, {256, 256, 256,  8},
      {256, 256, 512,  4}, {512, 512, 512,  4}, {512, 512, 512,  4},
      {512, 512, 512,  2}, {512, 512, 512,  2}, {512, 512, 512,  2}};
  static const bool poolAfter[13] = {false, true, false, true, false, false, true,
                                     false, false, true, false, false, true};

  const float* x = (const float*)d_in[0];

  // ---- carve workspace ----
  char* ws = (char*)d_ws;
  size_t off = 0;
  auto carve = [&](size_t bytes) -> char* {
    char* p = ws + off;
    off += (bytes + 255) & ~(size_t)255;
    return p;
  };

  bf16* bufI = (bf16*)carve((size_t)BATCH * 34 * 34 * 32 * sizeof(bf16));
  const size_t pingBytes = (size_t)BATCH * 34 * 34 * 64 * sizeof(bf16);  // largest layer
  bf16* bufs[2];
  bufs[0] = (bf16*)carve(pingBytes);
  bufs[1] = (bf16*)carve(pingBytes);
  bf16* wpk[13];
  for (int i = 0; i < 13; ++i)
    wpk[i] = (bf16*)carve((size_t)L[i].cout * L[i].cinPad * 9 * sizeof(bf16));

  // ---- pack weights (f32 -> bf16, B-fragment layout) ----
  for (int i = 0; i < 13; ++i) {
    const long total = (long)L[i].cout * L[i].cinPad * 9;
    const int grid = (int)((total + 255) / 256);
    pack_weight_kernel<<<grid, 256, 0, stream>>>(
        (const float*)d_in[1 + 2 * i], wpk[i], L[i].cin, L[i].cinPad, L[i].cout);
  }

  // ---- pack input (f32 NCHW -> padded channels-last bf16) ----
  pack_input_kernel<<<2048, 256, 0, stream>>>(x, bufI);

  // ---- layer pipeline (ping-pong) ----
  const bf16* cur = bufI;
  int nx = 0;
  for (int i = 0; i < 13; ++i) {
    const int hh = L[i].h;
    bf16* dst = bufs[nx];
    const long dstWords = (long)BATCH * (hh + 2) * (hh + 2) * L[i].cout / 2;
    zero_words_kernel<<<2048, 256, 0, stream>>>((uint32_t*)dst, dstWords);

    dim3 grid((BATCH * hh * hh) / 128, L[i].cout / 64);
    kerv2d_wmma_kernel<<<grid, 256, 0, stream>>>(
        cur, wpk[i], (const float*)d_in[2 + 2 * i], dst,
        hh, hh, L[i].cinPad, L[i].cout);
    cur = dst;
    nx ^= 1;

    if (poolAfter[i]) {
      bf16* pd = bufs[nx];
      avgpool2_kernel<<<2048, 256, 0, stream>>>(cur, pd, L[i].cout, hh);
      cur = pd;
      nx ^= 1;
    }
  }

  // ---- classifier ----
  fc_kernel<<<(BATCH * NUM_CLASSES + 255) / 256, 256, 0, stream>>>(
      cur, (const float*)d_in[27], (const float*)d_in[28], (float*)d_out);
}
